// ModulatedDeformConv2d_5729486373320
// MI455X (gfx1250) — matrix operand order repaired and checked by reading, not compiled, not run on hardware
//
#include <hip/hip_runtime.h>

typedef __attribute__((ext_vector_type(16))) __bf16 v16bf;
typedef __attribute__((ext_vector_type(8)))  float  v8f;
typedef unsigned int u32x4 __attribute__((ext_vector_type(4)));

// Problem constants (B=4, CIN=64, H=W=128, COUT=64, 3x3, stride=1, pad=1, dil=1)
static constexpr int kHW   = 16384;   // H*W
static constexpr int kWI   = 128;
static constexpr int kHI   = 128;
static constexpr int kCIN  = 64;
static constexpr int kCOUT = 64;
static constexpr int kK    = 9;
static constexpr int kQ    = 576;     // CIN*K (reduction dim)
static constexpr int kNPX  = 32;      // output pixels per block
static constexpr int kVSTR = 592;     // padded Vt row stride (bf16 elems), 1184B = 37*32B

__device__ __forceinline__ unsigned short f2bf(float f) {
    unsigned u = __builtin_bit_cast(unsigned, f);
    u += 0x7FFFu + ((u >> 16) & 1u);          // round-to-nearest-even
    return (unsigned short)(u >> 16);
}

// One-time weight f32 -> bf16 (64x576 row-major) into workspace.
__global__ void dcn_wcvt_kernel(const float* __restrict__ w,
                                unsigned short* __restrict__ wb, int n) {
    int i = blockIdx.x * blockDim.x + threadIdx.x;
    if (i < n) wb[i] = f2bf(w[i]);
}

__global__ __launch_bounds__(128)
void dcn_wmma_kernel(const float* __restrict__ x,
                     const float* __restrict__ off,
                     const float* __restrict__ msk,
                     const unsigned short* __restrict__ wbf,   // [COUT][Q] bf16
                     const float* __restrict__ bias,
                     float* __restrict__ out) {
    __shared__ __align__(32) unsigned short Vt[kNPX][kVSTR];   // v[q,px] stored as [px][q]

    const int tid  = threadIdx.x;
    const int p0   = blockIdx.x * kNPX;       // linear output index base over B*H*W
    const int b    = p0 >> 14;                // / kHW
    const int pimg = p0 & (kHW - 1);

    // ---------------- phase 1: deformable bilinear sampling -> Vt (bf16) ----------------
    for (int item = tid; item < kNPX * kK; item += 128) {
        const int px = item & (kNPX - 1);     // adjacent lanes -> adjacent pixels (coalesced)
        const int k  = item >> 5;
        const int pp = pimg + px;
        const int ho = pp >> 7;
        const int wo = pp & (kWI - 1);

        const float dy = off[((b * (2 * kK) + 2 * k    ) << 14) + pp];
        const float dx = off[((b * (2 * kK) + 2 * k + 1) << 14) + pp];
        const float mv = msk[((b * kK + k) << 14) + pp];

        const float py = dy + (float)(k / 3) + (float)(ho - 1);
        const float qx = dx + (float)(k % 3) + (float)(wo - 1);
        const float y0f = floorf(py), x0f = floorf(qx);
        const float wy = py - y0f,   wx = qx - x0f;
        const int y0 = (int)y0f, x0 = (int)x0f;
        const int y1 = y0 + 1,   x1 = x0 + 1;

        const float vy0 = (y0 >= 0 && y0 < kHI) ? 1.f : 0.f;
        const float vy1 = (y1 >= 0 && y1 < kHI) ? 1.f : 0.f;
        const float vx0 = (x0 >= 0 && x0 < kWI) ? 1.f : 0.f;
        const float vx1 = (x1 >= 0 && x1 < kWI) ? 1.f : 0.f;

        const float w00 = (1.f - wy) * (1.f - wx) * vy0 * vx0 * mv;
        const float w01 = (1.f - wy) * wx         * vy0 * vx1 * mv;
        const float w10 = wy * (1.f - wx)         * vy1 * vx0 * mv;
        const float w11 = wy * wx                 * vy1 * vx1 * mv;

        const int cy0 = min(max(y0, 0), kHI - 1), cy1 = min(max(y1, 0), kHI - 1);
        const int cx0 = min(max(x0, 0), kWI - 1), cx1 = min(max(x1, 0), kWI - 1);
        const int i00 = cy0 * kWI + cx0, i01 = cy0 * kWI + cx1;
        const int i10 = cy1 * kWI + cx0, i11 = cy1 * kWI + cx1;

        const float* xb = x + ((size_t)b * kCIN << 14);
        unsigned short* vrow = &Vt[px][k];
        #pragma unroll 4
        for (int c = 0; c < kCIN; ++c) {
            const float* xc = xb + ((size_t)c << 14);
            const float s = w00 * xc[i00] + w01 * xc[i01]
                          + w10 * xc[i10] + w11 * xc[i11];
            vrow[c * kK] = f2bf(s);          // q = c*9 + k, matches weight layout
        }
    }
    __syncthreads();

    // ---------------- phase 2: WMMA GEMM out[co,px] = W[co,q] * Vt[q,px] ----------------
    const int lane = tid & 31;
    const int wave = tid >> 5;
    const int coB  = wave * 16;                          // 4 waves cover COUT=64
    const int arow = coB + (lane & 15);                  // A: M = lane%16
    const int asel = (lane < 16) ? 0 : 16;               // A: K 0-15 vs 16-31
    const int bsel = (lane < 16) ? 0 : 16;               // B: K 0-15 vs 16-31
    const int bpx  = lane & 15;                          // B: N = lane%16

    v8f acc0 = {}; v8f acc1 = {};
    #pragma unroll 2
    for (int kc = 0; kc < kQ / 32; ++kc) {               // 18 chunks of K=32
        union { struct { u32x4 lo, hi; } u; v16bf v; } A;
        const unsigned short* wp = wbf + arow * kQ + kc * 32 + asel;
        A.u.lo = *(const u32x4*)(wp);                    // K s..s+7
        A.u.hi = *(const u32x4*)(wp + 8);                // K s+8..s+15

        const v16bf b0 = *(const v16bf*)&Vt[bpx     ][kc * 32 + bsel];
        const v16bf b1 = *(const v16bf*)&Vt[bpx + 16][kc * 32 + bsel];

        acc0 = __builtin_amdgcn_wmma_f32_16x16x32_bf16(false, A.v, false, b0,
                                                       (short)0, acc0, false, false);
        acc1 = __builtin_amdgcn_wmma_f32_16x16x32_bf16(false, A.v, false, b1,
                                                       (short)0, acc1, false, false);
    }

    // ---------------- epilogue: bias + store (C/D layout: VGPR r -> M=r / r+8) ----------
    const int hi8 = (lane < 16) ? 0 : 8;
    const int pxn = lane & 15;
    #pragma unroll
    for (int r = 0; r < 8; ++r) {
        const int co = coB + r + hi8;
        const float bv = bias[co];
        float* op = out + (((size_t)(b * kCOUT + co)) << 14) + pimg;
        op[pxn]      = acc0[r] + bv;
        op[pxn + 16] = acc1[r] + bv;
    }
}

extern "C" void kernel_launch(void* const* d_in, const int* in_sizes, int n_in,
                              void* d_out, int out_size, void* d_ws, size_t ws_size,
                              hipStream_t stream) {
    const float* x    = (const float*)d_in[0];
    const float* off  = (const float*)d_in[1];
    const float* msk  = (const float*)d_in[2];
    const float* wgt  = (const float*)d_in[3];
    const float* bias = (const float*)d_in[4];
    float* out = (float*)d_out;

    unsigned short* wbf = (unsigned short*)d_ws;         // 64*576 bf16 = 72 KB
    const int nw = kCOUT * kQ;
    dcn_wcvt_kernel<<<(nw + 255) / 256, 256, 0, stream>>>(wgt, wbf, nw);

    const int nblocks = (4 * kHW) / kNPX;                // 2048 blocks, 128 thr = 4 waves
    dcn_wmma_kernel<<<nblocks, 128, 0, stream>>>(x, off, msk, wbf, bias, out);
}
